// AttrDecoder_45466523796172
// MI455X (gfx1250) — matrix operand order repaired and checked by reading, not compiled, not run on hardware
//
#include <hip/hip_runtime.h>
#include <stdint.h>

typedef __attribute__((ext_vector_type(16))) __bf16          v16bf;
typedef __attribute__((ext_vector_type(16))) unsigned short  v16u;
typedef __attribute__((ext_vector_type(8)))  float           v8f;

__device__ __forceinline__ unsigned short f32_to_bf16(float f) {
    unsigned int u = __builtin_bit_cast(unsigned int, f);
    u += 0x7FFFu + ((u >> 16) & 1u);          // round-to-nearest-even
    return (unsigned short)(u >> 16);
}

// ---------------------------------------------------------------- zero fill
__global__ void zero_f32(float* __restrict__ p, long n) {
    long i = (long)blockIdx.x * blockDim.x + threadIdx.x;
    long stride = (long)gridDim.x * blockDim.x;
    for (; i < n; i += stride) p[i] = 0.0f;
}

// ---------------------------------------------------------------- degrees
__global__ void degree_kernel(const int* __restrict__ src, const int* __restrict__ dst,
                              float* __restrict__ deg_s, float* __restrict__ deg_d, int e) {
    int i = blockIdx.x * blockDim.x + threadIdx.x;
    if (i < e) {
        unsafeAtomicAdd(&deg_s[src[i]], 1.0f);   // global_atomic_add_f32
        unsafeAtomicAdd(&deg_d[dst[i]], 1.0f);
    }
}

// deg -> deg^-1/2 (0 if deg==0); both norm arrays are contiguous -> one pass
__global__ void invsqrt_kernel(float* __restrict__ p, int n) {
    int i = blockIdx.x * blockDim.x + threadIdx.x;
    if (i < n) {
        float d = p[i];
        p[i] = (d > 0.0f) ? rsqrtf(d) : 0.0f;
    }
}

// ------------------------------------------- edge scatter: agg[dst] += x[src]*norm_src[src]
// one thread per (edge, 4-feature block); float4 gather, 4 hw fp32 atomics scatter
__global__ void scatter_kernel(const float* __restrict__ x, const int* __restrict__ src,
                               const int* __restrict__ dst, const float* __restrict__ norm_s,
                               float* __restrict__ agg, int e, int din4) {
    int t = blockIdx.x * blockDim.x + threadIdx.x;
    int total = e * din4;
    if (t >= total) return;
    int ei = t / din4;
    int fb = t - ei * din4;
    int s = src[ei];
    int d = dst[ei];
    float sc = norm_s[s];
    const float4 v = reinterpret_cast<const float4*>(x + (long)s * din4 * 4)[fb];
    float* out = agg + (long)d * din4 * 4 + (long)fb * 4;
    unsafeAtomicAdd(out + 0, v.x * sc);
    unsafeAtomicAdd(out + 1, v.y * sc);
    unsafeAtomicAdd(out + 2, v.z * sc);
    unsafeAtomicAdd(out + 3, v.w * sc);
}

// ------------------------------------------- pack A: bf16( agg * norm_dst[:,None] ),
// pre-swizzled into the ISA 16-bit A 16x32 lane layout so the GEMM lane reads one
// contiguous 32B chunk (2 x global_load_b128) per K-step.
// layout: abuf[((mt*KSTEPS + ks)*32 + lane)] : v16u
__global__ void pack_a_kernel(const float* __restrict__ agg, const float* __restrict__ norm_d,
                              unsigned short* __restrict__ abuf, int n, int din) {
    int t = blockIdx.x * blockDim.x + threadIdx.x;
    int ksteps = din >> 5;
    int mtile  = (n + 15) >> 4;
    int total  = mtile * ksteps * 32;
    if (t >= total) return;
    int lane = t & 31;
    int rest = t >> 5;
    int ks   = rest % ksteps;
    int mt   = rest / ksteps;
    int half = lane >> 4, m = lane & 15;
    int row = mt * 16 + m;
    if (row > n - 1) row = n - 1;               // clamp (dup rows in pad tile, never stored)
    float sc = norm_d[row];
    const float* arp = agg + (long)row * din;
    int kk = ks << 5;
    v16u au;
#pragma unroll
    for (int v = 0; v < 8; ++v) {
        // A layout: K = (v>>2)*16 + half*8 + (v&3)*2  (pairs are K, K+1)
        int ka = kk + ((v >> 2) << 4) + (half << 3) + ((v & 3) << 1);
        float2 p = *reinterpret_cast<const float2*>(arp + ka);
        au[2 * v]     = f32_to_bf16(p.x * sc);
        au[2 * v + 1] = f32_to_bf16(p.y * sc);
    }
    reinterpret_cast<v16u*>(abuf)[t] = au;      // t == (mt*ksteps + ks)*32 + lane
}

// ------------------------------------------- pack B: bf16(W) pre-swizzled into the
// dense 16-bit B 32x16 layout.  layout: wbuf[((nt*KSTEPS + ks)*32 + lane)] : v16u
__global__ void pack_b_kernel(const float* __restrict__ W, unsigned short* __restrict__ wbuf,
                              int din, int dout) {
    int t = blockIdx.x * blockDim.x + threadIdx.x;
    int ksteps = din >> 5;
    int ntile  = dout >> 4;
    int total  = ntile * ksteps * 32;
    if (t >= total) return;
    int lane = t & 31;
    int rest = t >> 5;
    int ks   = rest % ksteps;
    int nt   = rest / ksteps;
    int half = lane >> 4, m = lane & 15;
    int col = nt * 16 + m;
    int kk  = ks << 5;
    v16u bu;
#pragma unroll
    for (int v = 0; v < 8; ++v) {
        // B layout: K = (v>>2)*16 + half*8 + (v&3)*2  (pairs are K, K+1), column = lane&15
        int kb = kk + ((v >> 2) << 4) + (half << 3) + ((v & 3) << 1);
        bu[2 * v]     = f32_to_bf16(W[(long)kb       * dout + col]);
        bu[2 * v + 1] = f32_to_bf16(W[(long)(kb + 1) * dout + col]);
    }
    reinterpret_cast<v16u*>(wbuf)[t] = bu;
}

// ------------------------------------------- y = relu(A_packed @ B_packed + b)
// one wave32 per 16x16 output tile; fully unrolled K loop of v_wmma_f32_16x16x32_bf16
template <int DIN>
__global__ void gemm_bias_relu_wmma(const unsigned short* __restrict__ abuf,
                                    const unsigned short* __restrict__ wbuf,
                                    const float* __restrict__ bias,
                                    float* __restrict__ y, int n, int dout) {
    constexpr int KSTEPS = DIN / 32;
    int wave  = (int)((blockIdx.x * blockDim.x + threadIdx.x) >> 5);
    int lane  = threadIdx.x & 31;
    int ntile = dout >> 4;
    int mtile = (n + 15) >> 4;
    if (wave >= mtile * ntile) return;          // wave-uniform: EXEC all-1s below
    int mt = wave / ntile;
    int nt = wave - mt * ntile;

    const v16u* ap = reinterpret_cast<const v16u*>(abuf) + ((long)mt * KSTEPS * 32 + lane);
    const v16u* bp = reinterpret_cast<const v16u*>(wbuf) + ((long)nt * KSTEPS * 32 + lane);

    v8f c = {};
#pragma unroll
    for (int ks = 0; ks < KSTEPS; ++ks) {
        v16bf a = __builtin_bit_cast(v16bf, ap[ks * 32]);
        v16bf b = __builtin_bit_cast(v16bf, bp[ks * 32]);
        // (neg_a, A, neg_b, B, c_mod, C, reuse_a, reuse_b)
        c = __builtin_amdgcn_wmma_f32_16x16x32_bf16(false, a, false, b, (short)0, c,
                                                    false, false);
    }

    int half = lane >> 4, m = lane & 15;
    int col  = nt * 16 + m;
    float bv = bias[col];
    // C/D layout: VGPR r holds M = r + half*8, N = lane&15
    float* yp = y + (long)(mt * 16 + (half << 3)) * dout + col;
    bool full = (mt * 16 + 16) <= n;            // wave-uniform fast path
    if (full) {
#pragma unroll
        for (int r = 0; r < 8; ++r) {
            float val = c[r] + bv;
            yp[(long)r * dout] = (val > 0.0f) ? val : 0.0f;
        }
    } else {
#pragma unroll
        for (int r = 0; r < 8; ++r) {
            int orow = mt * 16 + (half << 3) + r;
            if (orow < n) {
                float val = c[r] + bv;
                yp[(long)r * dout] = (val > 0.0f) ? val : 0.0f;
            }
        }
    }
}

static inline uintptr_t align_up(uintptr_t p, uintptr_t a) { return (p + a - 1) & ~(a - 1); }

// ---------------------------------------------------------------- driver
extern "C" void kernel_launch(void* const* d_in, const int* in_sizes, int n_in,
                              void* d_out, int out_size, void* d_ws, size_t ws_size,
                              hipStream_t stream) {
    const float* z   = (const float*)d_in[0];
    const int*   src = (const int*)d_in[1];
    const int*   dst = (const int*)d_in[2];
    const float* Ws[4] = {(const float*)d_in[3], (const float*)d_in[5],
                          (const float*)d_in[7], (const float*)d_in[9]};
    const float* bs[4] = {(const float*)d_in[4], (const float*)d_in[6],
                          (const float*)d_in[8], (const float*)d_in[10]};

    const int dims[5] = {32, 32, 64, 128, 128};
    const int n = in_sizes[0] / dims[0];   // 50000
    const int e = in_sizes[1];             // 800000
    const int n_pad = ((n + 15) / 16) * 16;

    float* ws      = (float*)d_ws;
    float* norm_s  = ws;                    // [n]
    float* norm_d  = ws + n;                // [n]
    float* agg     = norm_d + n;            // [n*128] f32
    float* bufA    = agg  + (long)n * 128;  // [n*128] f32
    float* bufB    = bufA + (long)n * 128;  // [n*128] f32
    unsigned short* abuf = (unsigned short*)align_up((uintptr_t)(bufB + (long)n * 128), 32);
    unsigned short* wbuf = (unsigned short*)align_up((uintptr_t)(abuf + (long)n_pad * 128), 32);

    // degrees -> symmetric norms
    zero_f32<<<1024, 256, 0, stream>>>(norm_s, 2L * n);
    degree_kernel<<<(e + 255) / 256, 256, 0, stream>>>(src, dst, norm_s, norm_d, e);
    invsqrt_kernel<<<(2 * n + 255) / 256, 256, 0, stream>>>(norm_s, 2 * n);

    const float* x = z;
    float* outs[4] = {bufA, bufB, bufA, (float*)d_out};
    for (int l = 0; l < 4; ++l) {
        int din  = dims[l];
        int dout = dims[l + 1];
        int ksteps = din / 32;
        int mtile  = (n + 15) / 16;
        int ntile  = dout / 16;

        zero_f32<<<2048, 256, 0, stream>>>(agg, (long)n * din);

        int din4 = din >> 2;
        long tot = (long)e * din4;
        scatter_kernel<<<(int)((tot + 255) / 256), 256, 0, stream>>>(
            x, src, dst, norm_s, agg, e, din4);

        int ta = mtile * ksteps * 32;
        pack_a_kernel<<<(ta + 255) / 256, 256, 0, stream>>>(agg, norm_d, abuf, n, din);
        int tb = ntile * ksteps * 32;
        pack_b_kernel<<<(tb + 255) / 256, 256, 0, stream>>>(Ws[l], wbuf, din, dout);

        int thr = mtile * ntile * 32;
        int blocks = (thr + 255) / 256;
        switch (din) {
        case 32:
            gemm_bias_relu_wmma<32><<<blocks, 256, 0, stream>>>(abuf, wbuf, bs[l], outs[l], n, dout);
            break;
        case 64:
            gemm_bias_relu_wmma<64><<<blocks, 256, 0, stream>>>(abuf, wbuf, bs[l], outs[l], n, dout);
            break;
        default:
            gemm_bias_relu_wmma<128><<<blocks, 256, 0, stream>>>(abuf, wbuf, bs[l], outs[l], n, dout);
            break;
        }
        x = outs[l];
    }
}
